// AttentionLayer_6700148982301
// MI455X (gfx1250) — compile-verified, matrix operand order checked by reading, not run on hardware
//
#include <hip/hip_runtime.h>
#include <hip/hip_bf16.h>

// ---------------------------------------------------------------------------
// AttentionLayer for MI455X (gfx1250): WMMA f16 GEMMs, fp32 accumulate.
//   B=2, S=2048, D=1024, H=16, Hd=64
// Outputs (concatenated in d_out): out [B,S,D] fp32, attn_mean [B,S,S] fp32.
// ---------------------------------------------------------------------------

#define B_  2
#define S_  2048
#define D_  1024
#define H_  16
#define HD_ 64
#define MROWS_ (B_ * S_)   // 4096

typedef __attribute__((ext_vector_type(16))) _Float16 v16h;
typedef __attribute__((ext_vector_type(8)))  float    v8f;

#define WMMA_F16(A, Bf, C) \
  __builtin_amdgcn_wmma_f32_16x16x32_f16(false, (A), false, (Bf), (short)0, (C), false, false)

// ---------------------------------------------------------------------------
// Load a 16-half WMMA fragment for a "row owner" lane from a row-major row.
// CDNA5 16-bit A-matrix 16x32 layout (ISA 7.12.2):
//   lane<16 : halves j=0..7 -> K = kb + j        ; j=8..15 -> K = kb + 16 + (j-8)
//   lane>=16: same + 8.
// Both runs are 8 contiguous halves -> two 16-byte loads.
// ---------------------------------------------------------------------------
__device__ __forceinline__ v16h load_frag_row(const _Float16* __restrict__ row,
                                              int kb, int hi) {
  union U { float4 f4; _Float16 h[8]; } u0, u1;
  u0.f4 = *(const float4*)(row + kb + hi * 8);
  u1.f4 = *(const float4*)(row + kb + 16 + hi * 8);
  v16h f;
#pragma unroll
  for (int j = 0; j < 8; ++j) { f[j] = u0.h[j]; f[8 + j] = u1.h[j]; }
  return f;
}

// ---------------------------------------------------------------------------
// Kernel 0a: x fp32 -> f16
// ---------------------------------------------------------------------------
__global__ __launch_bounds__(256) void cvt_x_kernel(const float* __restrict__ x,
                                                    _Float16* __restrict__ xh) {
  int i = (blockIdx.x * 256 + threadIdx.x) * 4;  // total 4096*1024, exact
  float4 v = *(const float4*)(x + i);
  xh[i + 0] = (_Float16)v.x;
  xh[i + 1] = (_Float16)v.y;
  xh[i + 2] = (_Float16)v.z;
  xh[i + 3] = (_Float16)v.w;
}

// ---------------------------------------------------------------------------
// Kernel 0b: transpose-convert Wq/Wk/Wv (fp32 [K][N]) -> f16 W^T [N][K].
// 32x32 LDS tile keeps both sides coalesced. z selects the weight.
// ---------------------------------------------------------------------------
__global__ __launch_bounds__(256) void cvt_wT_kernel(const float* __restrict__ Wq,
                                                     const float* __restrict__ Wk,
                                                     const float* __restrict__ Wv,
                                                     _Float16* __restrict__ wT) {
  __shared__ float t[32][33];
  const float* W = (blockIdx.z == 0) ? Wq : (blockIdx.z == 1) ? Wk : Wv;
  _Float16* dst = wT + (size_t)blockIdx.z * D_ * D_;
  int k0 = blockIdx.x * 32, n0 = blockIdx.y * 32;
  int tx = threadIdx.x & 31, ty = threadIdx.x >> 5;  // 8 rows per pass
#pragma unroll
  for (int r = ty; r < 32; r += 8)
    t[r][tx] = W[(size_t)(k0 + r) * D_ + n0 + tx];
  __syncthreads();
#pragma unroll
  for (int r = ty; r < 32; r += 8)
    dst[(size_t)(n0 + r) * D_ + k0 + tx] = (_Float16)t[tx][r];
}

// ---------------------------------------------------------------------------
// Kernel 1: QKV projection GEMM (M=4096, N=1024, K=1024) x3.
// Block = 8 waves arranged 4(M) x 2(N); each wave register-blocks a 32x32
// C tile (2x2 WMMA tiles) -> block tile 128x64. Per k-step: 2 A-frags +
// 2 B-frags feed 4 WMMAs (1 load_frag : 1 wmma).
// Q,K stored [B,H,S,Hd] f16; V stored transposed [B,H,Hd,S] f16 so the
// P@V B-fragments in the attention kernel are contiguous loads.
// ---------------------------------------------------------------------------
__global__ __launch_bounds__(256) void qkv_kernel(const _Float16* __restrict__ xh,
                                                  const _Float16* __restrict__ wT,
                                                  const float* __restrict__ bq,
                                                  const float* __restrict__ bk,
                                                  const float* __restrict__ bv,
                                                  _Float16* __restrict__ qh,
                                                  _Float16* __restrict__ kh,
                                                  _Float16* __restrict__ vh) {
  const int lane = threadIdx.x & 31;
  const int wave = threadIdx.x >> 5;
  const int hi = lane >> 4, ln = lane & 15;
  const int m0 = blockIdx.x * 128 + (wave >> 1) * 32;  // 4 M-waves
  const int n0 = blockIdx.y * 64 + (wave & 1) * 32;    // 2 N-waves
  const int z  = blockIdx.z;

  const _Float16* wz   = wT + (size_t)z * D_ * D_;
  const float*    bias = (z == 0) ? bq : (z == 1) ? bk : bv;
  const _Float16* ar0 = xh + (size_t)(m0 + ln) * D_;
  const _Float16* ar1 = xh + (size_t)(m0 + 16 + ln) * D_;
  const _Float16* br0 = wz + (size_t)(n0 + ln) * D_;
  const _Float16* br1 = wz + (size_t)(n0 + 16 + ln) * D_;

  v8f c00 = {}, c01 = {}, c10 = {}, c11 = {};
  for (int kb = 0; kb < D_; kb += 32) {
    v16h a0 = load_frag_row(ar0, kb, hi);
    v16h a1 = load_frag_row(ar1, kb, hi);
    v16h b0 = load_frag_row(br0, kb, hi);
    v16h b1 = load_frag_row(br1, kb, hi);
    c00 = WMMA_F16(a0, b0, c00);
    c01 = WMMA_F16(a0, b1, c01);
    c10 = WMMA_F16(a1, b0, c10);
    c11 = WMMA_F16(a1, b1, c11);
  }

#pragma unroll
  for (int mi = 0; mi < 2; ++mi) {
#pragma unroll
    for (int ni = 0; ni < 2; ++ni) {
      const v8f& c = (mi == 0) ? (ni == 0 ? c00 : c01) : (ni == 0 ? c10 : c11);
      const int n = n0 + ni * 16 + ln;
      const float bn = bias[n];
      const int hh = n >> 6, hd = n & 63;
#pragma unroll
      for (int r = 0; r < 8; ++r) {
        int m = m0 + mi * 16 + r + hi * 8;  // C layout: lane<16->M=r else r+8
        int bb = m >> 11, s = m & (S_ - 1);
        _Float16 val = (_Float16)(c[r] + bn);
        size_t bh = (size_t)(bb * H_ + hh);
        if (z == 0)      qh[(bh * S_ + s) * HD_ + hd] = val;
        else if (z == 1) kh[(bh * S_ + s) * HD_ + hd] = val;
        else             vh[(bh * HD_ + hd) * S_ + s] = val;  // V transposed
      }
    }
  }
}

// ---------------------------------------------------------------------------
// Kernel 2: fused attention. One block per (b, 16-query tile); loops all 16
// heads so the head-mean accumulates into exclusively-owned global rows
// (deterministic, no float atomics). Full 16x2048 score tile in LDS.
// P@V: each wave takes 1/8 of K for ALL 4 n-tiles (one A-fragment build
// feeds 4 WMMAs); partials reduced through the (dead) score buffer.
// ---------------------------------------------------------------------------
__global__ __launch_bounds__(256) void attn_kernel(const _Float16* __restrict__ qh,
                                                   const _Float16* __restrict__ kh,
                                                   const _Float16* __restrict__ vh,
                                                   float* __restrict__ ctx,
                                                   float* __restrict__ attn_mean) {
  __shared__ float sP[16][S_ + 1];  // 131 KB of the 320 KB WGP LDS; padded

  const int lane = threadIdx.x & 31;
  const int wave = threadIdx.x >> 5;
  const int hi = lane >> 4, ln = lane & 15;
  const int b  = blockIdx.x >> 7;          // /128 q-tiles per batch
  const int q0 = (blockIdx.x & 127) << 4;  // *16

  for (int h = 0; h < H_; ++h) {
    const size_t bh = (size_t)(b * H_ + h);
    const _Float16* qb  = qh + bh * S_ * HD_;
    const _Float16* kb_ = kh + bh * S_ * HD_;
    const _Float16* vtb = vh + bh * HD_ * S_;   // [Hd][S]

    // Q fragments: one A operand reused across all 16 key tiles of the wave
    const _Float16* qrow = qb + (size_t)(q0 + ln) * HD_;
    v16h a0 = load_frag_row(qrow, 0, hi);
    v16h a1 = load_frag_row(qrow, 32, hi);

    __syncthreads();  // previous head finished reading sP scratch

    // --- scores: wave w covers key columns [w*256, w*256+256) ---
    for (int kt = 0; kt < 16; ++kt) {
      int n0k = wave * 256 + kt * 16;
      const _Float16* krow = kb_ + (size_t)(n0k + ln) * HD_;
      v16h b0 = load_frag_row(krow, 0, hi);
      v16h b1 = load_frag_row(krow, 32, hi);
      v8f c = {};
      c = WMMA_F16(a0, b0, c);
      c = WMMA_F16(a1, b1, c);
#pragma unroll
      for (int r = 0; r < 8; ++r)
        sP[r + hi * 8][n0k + ln] = c[r] * 0.125f;  // 1/sqrt(64)
    }
    __syncthreads();

    // --- softmax: wave w owns rows 2w, 2w+1; mean-over-heads to d_out ---
    float* mbase = attn_mean + ((size_t)b * S_ + q0) * S_;
    for (int rr = 0; rr < 2; ++rr) {
      int r = wave * 2 + rr;
      float m = -3.0e38f;
      for (int j = lane; j < S_; j += 32) m = fmaxf(m, sP[r][j]);
#pragma unroll
      for (int off = 16; off; off >>= 1) m = fmaxf(m, __shfl_xor(m, off, 32));
      float sum = 0.f;
      for (int j = lane; j < S_; j += 32) {
        float e = __expf(sP[r][j] - m);
        sP[r][j] = e;
        sum += e;
      }
#pragma unroll
      for (int off = 16; off; off >>= 1) sum += __shfl_xor(sum, off, 32);
      float rs = 1.0f / sum;
      float* mrow = mbase + (size_t)r * S_;
      for (int j = lane; j < S_; j += 32) {
        float p = sP[r][j] * rs;
        sP[r][j] = p;
        float contrib = p * (1.0f / (float)H_);
        if (h == 0) mrow[j] = contrib;       // first head overwrites poison
        else        mrow[j] += contrib;      // block exclusively owns rows
      }
    }
    __syncthreads();

    // --- ctx = P @ V : 8-way split-K; each wave does all 4 n-tiles ---
    v8f c0 = {}, c1 = {}, c2 = {}, c3 = {};
    const _Float16* vr = vtb + (size_t)ln * S_;
    for (int ck = wave; ck < S_ / 32; ck += 8) {
      int kk0 = ck * 32;
      v16h a;
#pragma unroll
      for (int j = 0; j < 16; ++j) {
        int kk = kk0 + j + ((j & 8) ? 8 : 0) + hi * 8;  // A-frag K mapping
        a[j] = (_Float16)sP[ln][kk];
      }
      v16h bf0 = load_frag_row(vr,            kk0, hi);
      v16h bf1 = load_frag_row(vr + 16 * S_,  kk0, hi);
      v16h bf2 = load_frag_row(vr + 32 * S_,  kk0, hi);
      v16h bf3 = load_frag_row(vr + 48 * S_,  kk0, hi);
      c0 = WMMA_F16(a, bf0, c0);
      c1 = WMMA_F16(a, bf1, c1);
      c2 = WMMA_F16(a, bf2, c2);
      c3 = WMMA_F16(a, bf3, c3);
    }
    __syncthreads();  // all P reads done -> score buffer becomes scratch

    // stash partials: scratch[(wave*4 + nt)*256 + M*16 + col], 32 KB
    float* scratch = &sP[0][0];
    {
      const int base = wave * 4 * 256;
#pragma unroll
      for (int r = 0; r < 8; ++r) {
        int M16 = (r + hi * 8) * 16 + ln;
        scratch[base + 0 * 256 + M16] = c0[r];
        scratch[base + 1 * 256 + M16] = c1[r];
        scratch[base + 2 * 256 + M16] = c2[r];
        scratch[base + 3 * 256 + M16] = c3[r];
      }
    }
    __syncthreads();

    // elementwise reduce across the 8 wave partials; 4 outputs per thread
    {
      const int e0 = threadIdx.x * 4;   // 0..1023 over the 16x64 ctx tile
      const int M = e0 >> 6;
      const int n = e0 & 63;
      const int nt = n >> 4, col = n & 15;
      float a0r = 0.f, a1r = 0.f, a2r = 0.f, a3r = 0.f;
#pragma unroll
      for (int w = 0; w < 8; ++w) {
        const float* p = scratch + (w * 4 + nt) * 256 + M * 16 + col;
        a0r += p[0]; a1r += p[1]; a2r += p[2]; a3r += p[3];
      }
      float4 o; o.x = a0r; o.y = a1r; o.z = a2r; o.w = a3r;
      *(float4*)(ctx + (bh * S_ + (size_t)(q0 + M)) * HD_ + n) = o;
    }
  }
}

// ---------------------------------------------------------------------------
// Kernel 3: out = LayerNorm(x + ctx) * gamma + beta. One block per row.
// ---------------------------------------------------------------------------
__global__ __launch_bounds__(256) void ln_kernel(const float* __restrict__ x,
                                                 const float* __restrict__ ctx,
                                                 const float* __restrict__ gamma,
                                                 const float* __restrict__ beta,
                                                 float* __restrict__ out) {
  __shared__ float red1[8], red2[8];
  const int row = blockIdx.x;             // 0..4095
  const int b = row >> 11, s = row & (S_ - 1);
  const int t = threadIdx.x, lane = t & 31, wave = t >> 5;
  const int d0 = t * 4;
  const int hh = d0 >> 6, hd = d0 & 63;

  float4 xv = *(const float4*)(x + (size_t)row * D_ + d0);
  float4 cv = *(const float4*)(ctx + (((size_t)(b * H_ + hh) * S_ + s) * HD_ + hd));
  float y0 = xv.x + cv.x, y1 = xv.y + cv.y, y2 = xv.z + cv.z, y3 = xv.w + cv.w;
  float s1 = y0 + y1 + y2 + y3;
  float s2 = y0 * y0 + y1 * y1 + y2 * y2 + y3 * y3;
#pragma unroll
  for (int off = 16; off; off >>= 1) {
    s1 += __shfl_xor(s1, off, 32);
    s2 += __shfl_xor(s2, off, 32);
  }
  if (lane == 0) { red1[wave] = s1; red2[wave] = s2; }
  __syncthreads();
  if (wave == 0) {
    float a = (lane < 8) ? red1[lane] : 0.f;
    float c = (lane < 8) ? red2[lane] : 0.f;
#pragma unroll
    for (int off = 16; off; off >>= 1) {
      a += __shfl_xor(a, off, 32);
      c += __shfl_xor(c, off, 32);
    }
    if (lane == 0) { red1[0] = a; red2[0] = c; }
  }
  __syncthreads();
  const float mu  = red1[0] * (1.0f / (float)D_);
  const float var = red2[0] * (1.0f / (float)D_) - mu * mu;
  const float rs  = rsqrtf(var + 1e-5f);
  float4 gv = *(const float4*)(gamma + d0);
  float4 bv = *(const float4*)(beta + d0);
  float4 o;
  o.x = (y0 - mu) * rs * gv.x + bv.x;
  o.y = (y1 - mu) * rs * gv.y + bv.y;
  o.z = (y2 - mu) * rs * gv.z + bv.z;
  o.w = (y3 - mu) * rs * gv.w + bv.w;
  *(float4*)(out + (size_t)row * D_ + d0) = o;
}

// ---------------------------------------------------------------------------
extern "C" void kernel_launch(void* const* d_in, const int* in_sizes, int n_in,
                              void* d_out, int out_size, void* d_ws, size_t ws_size,
                              hipStream_t stream) {
  (void)in_sizes; (void)n_in; (void)out_size; (void)ws_size;
  const float* x     = (const float*)d_in[0];
  const float* Wq    = (const float*)d_in[1];
  const float* bq    = (const float*)d_in[2];
  const float* Wk    = (const float*)d_in[3];
  const float* bk    = (const float*)d_in[4];
  const float* Wv    = (const float*)d_in[5];
  const float* bv    = (const float*)d_in[6];
  const float* gamma = (const float*)d_in[7];
  const float* beta  = (const float*)d_in[8];

  float* out       = (float*)d_out;
  float* attn_mean = out + (size_t)B_ * S_ * D_;  // second output, flat concat

  // Workspace layout (total ~54 MB):
  _Float16* xh = (_Float16*)d_ws;                       // [4096][1024] f16
  _Float16* wT = xh + (size_t)MROWS_ * D_;              // 3 x [1024][1024] f16
  _Float16* qh = wT + (size_t)3 * D_ * D_;              // [B,H,S,Hd] f16
  _Float16* kh = qh + (size_t)MROWS_ * D_;              // [B,H,S,Hd] f16
  _Float16* vh = kh + (size_t)MROWS_ * D_;              // [B,H,Hd,S] f16
  float*   ctx = (float*)(vh + (size_t)MROWS_ * D_);    // [B,H,S,Hd] f32

  cvt_x_kernel<<<(MROWS_ * D_) / (256 * 4), 256, 0, stream>>>(x, xh);
  cvt_wT_kernel<<<dim3(D_ / 32, D_ / 32, 3), 256, 0, stream>>>(Wq, Wk, Wv, wT);
  qkv_kernel<<<dim3(MROWS_ / 128, D_ / 64, 3), 256, 0, stream>>>(
      xh, wT, bq, bk, bv, qh, kh, vh);
  attn_kernel<<<B_ * (S_ / 16), 256, 0, stream>>>(qh, kh, vh, ctx, attn_mean);
  ln_kernel<<<MROWS_, 256, 0, stream>>>(x, ctx, gamma, beta, out);
}
